// HebbianConvTranspose2d_88158498718405
// MI455X (gfx1250) — matrix operand order repaired and checked by reading, not compiled, not run on hardware
//
#include <hip/hip_runtime.h>
#include <hip/hip_bf16.h>
#include <stdint.h>

// ---------------------------------------------------------------------------
// Transposed conv (stride 2, k=4, pad 1):  x(16,256,64,64) f32 -> out(16,128,128,128) f32
// Implicit GEMM per (batch, y, ph, pw):  D[128 x 64] = A[128 x 1024] * B[1024 x 64]
//   K index = c*4 + th*2 + tw ;  kh = 2*th+ph ; ih = y+th+ph-1 (same for width)
// bf16 WMMA (v_wmma_f32_16x16x32_bf16), f32 accumulation, double-buffered LDS,
// async global->LDS staging (global_load_async_to_lds_b128) for weight tiles.
// ---------------------------------------------------------------------------

typedef __attribute__((ext_vector_type(16))) __bf16 v16bf;
typedef __attribute__((ext_vector_type(8)))  float  v8f;

#define O_CH   128
#define C_CH   256
#define HIN    64
#define WIN    64
#define HOUT   128
#define WOUT   128
#define KCHUNKS 32          // 1024 / 32
#define ASTRIDE 40          // padded LDS row stride (elements): 80B -> conflict-free b128 reads
#define BSTRIDE 40
#define A_ELEMS (O_CH * ASTRIDE)   // 5120 ushorts per buffer
#define B_ELEMS (64   * BSTRIDE)   // 2560 ushorts per buffer

#if defined(__gfx1250__) && defined(__HIP_DEVICE_COMPILE__)
#if __has_builtin(__builtin_amdgcn_global_load_async_to_lds_b128) && \
    __has_builtin(__builtin_amdgcn_s_wait_asynccnt)
#define USE_ASYNC_LDS 1
#endif
#endif

#ifdef USE_ASYNC_LDS
// Exact parameter type per hipcc diagnostic:
//   param0: __attribute__((vector_size(16))) int  __device__ * (addrspace(1))
//   param1: LDS-side pointer (addrspace(3))
typedef int v4i __attribute__((vector_size(16)));
typedef __attribute__((address_space(1))) v4i* gv4i_p;
typedef __attribute__((address_space(3))) v4i* lv4i_p;
#endif

__device__ __forceinline__ unsigned short f2bf(float f) {
    union { float f; uint32_t u; } v; v.f = f;
    uint32_t u = v.u;
    return (unsigned short)((u + 0x7FFFu + ((u >> 16) & 1u)) >> 16);  // RNE
}

union FragAB { uint4 q[2]; v16bf v; };

// ---------------------------------------------------------------------------
// Weight prepack: w[o][c*16 + kh*4 + kw] f32 -> Wp[((p*32+kc)*128 + o)*32 + kk] bf16
//   p = ph*2+pw ; k = kc*32+kk = c*4 + th*2 + tw ; kh = 2*th+ph ; kw = 2*tw+pw
// Each (p,kc) block is a contiguous 8KB chunk for coalesced / async staging.
// ---------------------------------------------------------------------------
__global__ void prepack_w_kernel(const float* __restrict__ w,
                                 unsigned short* __restrict__ wp) {
    int idx = blockIdx.x * 256 + threadIdx.x;   // 0 .. 524287
    int kk = idx & 31;
    int o  = (idx >> 5)  & 127;
    int kc = (idx >> 12) & 31;
    int p  =  idx >> 17;
    int ph = p >> 1, pw = p & 1;
    int k  = kc * 32 + kk;
    int c  = k >> 2;
    int th = (k >> 1) & 1;
    int tw =  k       & 1;
    int kh = 2 * th + ph;
    int kw = 2 * tw + pw;
    wp[idx] = f2bf(w[o * 4096 + c * 16 + kh * 4 + kw]);
}

// ---------------------------------------------------------------------------
// Main kernel: one block per (n, y, ph, pw).  256 threads = 8 waves (wave32).
// Wave w computes out-channels [16w,16w+16) x 64 pixels via 4 WMMA tiles.
// ---------------------------------------------------------------------------
__global__ void __launch_bounds__(256)
tconv_wmma_kernel(const float* __restrict__ x,
                  const unsigned short* __restrict__ wp,
                  float* __restrict__ out) {
    __shared__ unsigned short sA[2 * A_ELEMS];   // weights  [buf][o][kk]
    __shared__ unsigned short sB[2 * B_ELEMS];   // patches  [buf][pixel][kk]

    int b  = blockIdx.x;          // 0 .. 4095
    int pw = b & 1;
    int ph = (b >> 1) & 1;
    int y  = (b >> 2) & 63;
    int n  = b >> 8;
    int p  = ph * 2 + pw;

    int tid    = threadIdx.x;
    int wave   = tid >> 5;
    int lane   = tid & 31;
    int laneM  = lane & 15;
    int laneHi = lane >> 4;       // 0 or 1

    const unsigned short* wpp = wp + (size_t)p * (32u * 128u * 32u);
    const float* xb = x + (size_t)n * C_CH * HIN * WIN;

    // ---- Hoisted, K-chunk-invariant staging state ---------------------------
    // A: two 16B chunks per thread per stage (512 x 16B per chunk)
    const int ach0  = tid;
    const int ach1  = tid + 256;
    const int alds0 = (ach0 >> 2) * ASTRIDE + (ach0 & 3) * 8;
    const int alds1 = (ach1 >> 2) * ASTRIDE + (ach1 & 3) * 8;
    // B: 8 gathered elements per thread; only the channel base moves with kc.
    int  boff[8];   // index into xb for chunk 0 (clamped to 0 when OOB)
    bool bval[8];
    int  blds[8];
    #pragma unroll
    for (int r = 0; r < 8; ++r) {
        int idx = r * 256 + tid;      // 0 .. 2047 ; idx = kk*64 + j (coalesced iw)
        int kk  = idx >> 6;
        int j   = idx & 63;
        int cc  = kk >> 2;
        int th  = (kk >> 1) & 1;
        int tw  =  kk       & 1;
        int ih  = y + th + ph - 1;
        int iw  = j + tw + pw - 1;
        bval[r] = ((unsigned)ih < (unsigned)HIN) && ((unsigned)iw < (unsigned)WIN);
        boff[r] = bval[r] ? ((cc * HIN + ih) * WIN + iw) : 0;
        blds[r] = j * BSTRIDE + kk;
    }
    // Fragment read offsets (ISA WMMA VGPR layouts), invariant:
    const int aoff = (wave * 16 + laneM) * ASTRIDE + laneHi * 8;
    int bfoff[4];
    #pragma unroll
    for (int nt = 0; nt < 4; ++nt)
        bfoff[nt] = (nt * 16 + laneM) * BSTRIDE + laneHi * 8;

    v8f acc0 = {}, acc1 = {}, acc2 = {}, acc3 = {};

    // ---- Stage one K-chunk into buffer `buf` --------------------------------
    auto stage = [&](int buf, int kcn) {
        unsigned short* sAb = sA + buf * A_ELEMS;
        unsigned short* sBb = sB + buf * B_ELEMS;
        const unsigned short* asrc = wpp + kcn * (128 * 32);   // contiguous 8KB
#ifdef USE_ASYNC_LDS
        __builtin_amdgcn_global_load_async_to_lds_b128(
            (gv4i_p)(asrc + ach0 * 8), (lv4i_p)&sAb[alds0], 0, 0);
        __builtin_amdgcn_global_load_async_to_lds_b128(
            (gv4i_p)(asrc + ach1 * 8), (lv4i_p)&sAb[alds1], 0, 0);
#else
        *(uint4*)&sAb[alds0] = *(const uint4*)(asrc + ach0 * 8);
        *(uint4*)&sAb[alds1] = *(const uint4*)(asrc + ach1 * 8);
#endif
        const float* xc = xb + (size_t)kcn * (8 * HIN * WIN);
        #pragma unroll
        for (int r = 0; r < 8; ++r) {
            float v = xc[boff[r]];
            v = bval[r] ? v : 0.0f;
            sBb[blds[r]] = f2bf(v);
        }
    };

    // ---- Software-pipelined main loop (double-buffered LDS) -----------------
    stage(0, 0);
    for (int kc = 0; kc < KCHUNKS; ++kc) {
        int cur = kc & 1;
#ifdef USE_ASYNC_LDS
        __builtin_amdgcn_s_wait_asynccnt(0);
#endif
        __syncthreads();                       // buffer `cur` ready; prev compute done
        if (kc + 1 < KCHUNKS) stage(cur ^ 1, kc + 1);

        const unsigned short* sAc = sA + cur * A_ELEMS;
        const unsigned short* sBc = sB + cur * B_ELEMS;

        FragAB a;
        a.q[0] = *(const uint4*)(sAc + aoff);        // k = kb .. kb+7
        a.q[1] = *(const uint4*)(sAc + aoff + 16);   // k = kb+16 .. kb+23
        #pragma unroll
        for (int nt = 0; nt < 4; ++nt) {
            FragAB bfr;
            bfr.q[0] = *(const uint4*)(sBc + bfoff[nt]);      // k = kb .. kb+7
            bfr.q[1] = *(const uint4*)(sBc + bfoff[nt] + 16); // k = kb+16 .. kb+23
            v8f* accp = (nt == 0) ? &acc0 : (nt == 1) ? &acc1 : (nt == 2) ? &acc2 : &acc3;
#if defined(__gfx1250__)
            *accp = __builtin_amdgcn_wmma_f32_16x16x32_bf16(
                false, a.v, false, bfr.v, (short)0, *accp, false, false);
#else
            (void)bfr; (void)accp;
#endif
        }
    }

    // ---- Store: C/D layout -> lane L: n = L%16, m = 8*(L/16)+v --------------
    int oh = 2 * y + ph;
    float* ob = out + (size_t)n * O_CH * HOUT * WOUT;
    v8f accs[4] = {acc0, acc1, acc2, acc3};
    #pragma unroll
    for (int nt = 0; nt < 4; ++nt) {
        int j  = nt * 16 + laneM;
        int ow = 2 * j + pw;
        #pragma unroll
        for (int v = 0; v < 8; ++v) {
            int o = wave * 16 + laneHi * 8 + v;
            ob[((size_t)o * HOUT + oh) * WOUT + ow] = accs[nt][v];
        }
    }
}

// ---------------------------------------------------------------------------
extern "C" void kernel_launch(void* const* d_in, const int* in_sizes, int n_in,
                              void* d_out, int out_size, void* d_ws, size_t ws_size,
                              hipStream_t stream) {
    (void)in_sizes; (void)n_in; (void)out_size; (void)ws_size;
    const float* x = (const float*)d_in[0];          // (16,256,64,64) f32
    const float* w = (const float*)d_in[1];          // (128,4096)     f32
    float* out = (float*)d_out;                      // (16,128,128,128) f32
    unsigned short* wp = (unsigned short*)d_ws;      // 1 MB packed bf16 weights

    // 4*32*128*32 = 524288 elements -> 2048 blocks of 256
    prepack_w_kernel<<<2048, 256, 0, stream>>>(w, wp);
    // 16 batch * 64 y * 2 ph * 2 pw = 4096 blocks
    tconv_wmma_kernel<<<4096, 256, 0, stream>>>(x, wp, out);
}
